// DeformConvTranspose1d_11759620457159
// MI455X (gfx1250) — compile-verified, matrix operand order checked by reading, not run on hardware
//
#include <hip/hip_runtime.h>

// ---------------- problem constants (from reference) ----------------
#define B_    8
#define CIN   256
#define WIN   4096
#define COUT  256
#define KW    4
#define WOUT  8192      // (4096-1)*2 - 2 + 3 + 0 + 1
// stride=2, pad=1, dil=1, groups=1, offset_groups=1

typedef __attribute__((ext_vector_type(16))) __bf16 v16bf;
typedef __attribute__((ext_vector_type(8)))  float  v8f;

union V16U { v16bf v; uint4 u[2]; };

__device__ __forceinline__ unsigned short f32_to_bf16_rne(float f) {
    unsigned u = __float_as_uint(f);
    u += 0x7FFFu + ((u >> 16) & 1u);       // round-to-nearest-even
    return (unsigned short)(u >> 16);
}

// ---------------------------------------------------------------
// Kernel 1: repack weight [ci, co, k] f32  ->  Wb[k][co][ci] bf16
// ---------------------------------------------------------------
__global__ void weight_prep(const float* __restrict__ w,
                            unsigned short* __restrict__ wb) {
    int idx = blockIdx.x * blockDim.x + threadIdx.x;   // (k,co,ci), 2^18 total
    int ci = idx & 255;
    int co = (idx >> 8) & 255;
    int k  = idx >> 16;
    float f = w[((size_t)ci * COUT + co) * KW + k];
    wb[idx] = f32_to_bf16_rne(f);
}

// ---------------------------------------------------------------
// Kernel 2: out[b, co, w] = bias[co]
// ---------------------------------------------------------------
__global__ void out_init(float* __restrict__ out, const float* __restrict__ bias) {
    int idx = blockIdx.x * blockDim.x + threadIdx.x;
    int co = (idx >> 13) & 255;            // WOUT = 8192 = 1<<13
    out[idx] = bias[co];
}

// ---------------------------------------------------------------
// Kernel 3: fused bf16-WMMA GEMM + modulation + deformable lerp
//           scatter. offset/mask staged once per WG via
//           global_load_async_to_lds (ASYNCcnt); scatter accumulates
//           into an LDS window with ds_add_f32; only the window flush
//           (and rare offset outliers) touch global atomics.
//
// block = 512 threads (16 waves). blockIdx = (i-block of 32, batch b).
// Wave w owns co-tile [16w, 16w+16) x i-tile [i0, i0+32), all 4 taps.
// ---------------------------------------------------------------
#define LDS_STRIDE 264     // x tile row: 256 ci + pad (b128-aligned chunks)
#define WSPAN      96      // LDS scatter window: positions [2*i0-16, 2*i0+80)
#define WPITCH     97      // padded row pitch (floats) -> conflict-free l/l+16
#define WLO        16      // window starts at 2*i0 - WLO

__launch_bounds__(512)
__global__ void deform_convT_fused(const float* __restrict__ x,
                                   const unsigned short* __restrict__ wb,
                                   const float* __restrict__ offset,
                                   const float* __restrict__ mask,
                                   float* __restrict__ out) {
    __shared__ __align__(16) unsigned short Xs[32 * LDS_STRIDE];   // 16.5 KB
    __shared__ float win[COUT * WPITCH];                            // 99.3 KB
    __shared__ float om_s[256];   // [0..127]=offset(k,ii), [128..255]=mask(k,ii)

    const int t  = threadIdx.x;
    const int b  = blockIdx.y;
    const int i0 = blockIdx.x * 32;
    const int base_p = 2 * i0 - WLO;

    __builtin_prefetch(wb, 0, 1);          // global_prefetch_b8 on weight stream

    // ---- async-stage offset/mask tile (1 dword/thread, ASYNCcnt) ----
    if (t < 256) {
        const int local = t & 127;
        const int k  = local >> 5;
        const int ii = local & 31;
        const float* src = (t < 128 ? offset : mask) +
                           ((size_t)(b * KW + k) * WIN + i0 + ii);
        unsigned lds_addr = (unsigned)(size_t)&om_s[t];   // low 32b = LDS offset
        asm volatile("global_load_async_to_lds_b32 %0, %1, off"
                     :: "v"(lds_addr), "v"(src)
                     : "memory");
    }

    // ---- zero the scatter window ----
    for (int j = t; j < COUT * WPITCH; j += 512)
        win[j] = 0.0f;

    // ---- stage x tile [256 ci][32 i] -> LDS transposed bf16 [i][ci] ----
    const float* xb = x + (size_t)b * CIN * WIN + i0;
    #pragma unroll
    for (int e = 0; e < 8; ++e) {
        int idx = e * 512 + t;             // 4096 float2's: ci = idx>>4, pair = idx&15
        int ci = idx >> 4;
        int ii = (idx & 15) * 2;
        float2 f2 = *(const float2*)(xb + (size_t)ci * WIN + ii);
        Xs[ii * LDS_STRIDE + ci]       = f32_to_bf16_rne(f2.x);
        Xs[(ii + 1) * LDS_STRIDE + ci] = f32_to_bf16_rne(f2.y);
    }
    asm volatile("s_wait_asynccnt 0" ::: "memory");   // async ops landed in LDS
    __syncthreads();

    const int lane   = t & 31;
    const int wave   = t >> 5;             // 0..15
    const int cobase = wave * 16;
    const int half   = lane >> 4;          // 0 | 1
    const int lrow   = lane & 15;          // A row (co) / B,C col (i)
    const int chunk  = half * 8;           // 16-bit operand K-chunk base

    v8f acc[4][2];
    #pragma unroll
    for (int k = 0; k < 4; ++k) {
        acc[k][0] = {};
        acc[k][1] = {};
    }

    // ---- GEMM: ci in steps of 32; B reused across the 4 taps ----
    #pragma unroll
    for (int ci0 = 0; ci0 < CIN; ci0 += 32) {
        V16U b0, b1;
        {
            const unsigned short* p = &Xs[lrow * LDS_STRIDE + ci0 + chunk];
            b0.u[0] = *(const uint4*)(p);
            b0.u[1] = *(const uint4*)(p + 16);
            const unsigned short* q = &Xs[(16 + lrow) * LDS_STRIDE + ci0 + chunk];
            b1.u[0] = *(const uint4*)(q);
            b1.u[1] = *(const uint4*)(q + 16);
        }
        #pragma unroll
        for (int k = 0; k < 4; ++k) {
            V16U a;
            const unsigned short* pw =
                wb + (((size_t)(k * 256 + cobase + lrow)) * 256 + ci0 + chunk);
            a.u[0] = *(const uint4*)(pw);
            a.u[1] = *(const uint4*)(pw + 16);
            acc[k][0] = __builtin_amdgcn_wmma_f32_16x16x32_bf16(
                false, a.v, false, b0.v, (short)0, acc[k][0], false, false);
            acc[k][1] = __builtin_amdgcn_wmma_f32_16x16x32_bf16(
                false, a.v, false, b1.v, (short)0, acc[k][1], false, false);
        }
    }

    // ---- epilogue: modulation + lerp scatter into LDS window ----
    // C/D layout: col = lane&15, row = (lane>>4)*8 + r
    const int corow = cobase + half * 8;
    float* obase0 = out + ((size_t)b * COUT + corow) * WOUT;
    #pragma unroll
    for (int n = 0; n < 2; ++n) {
        const int i = i0 + n * 16 + lrow;
        #pragma unroll
        for (int k = 0; k < 4; ++k) {
            const float off = om_s[k * 32 + n * 16 + lrow];
            const float m   = om_s[128 + k * 32 + n * 16 + lrow];
            const float pos = off + (float)(i * 2 - 1 + k);   // i*stride - pad + k*dil
            const float p0f = floorf(pos);
            const float fr  = pos - p0f;
            const int   p0  = (int)p0f;
            const float w0  = (1.0f - fr) * m;
            const float w1  = fr * m;
            const int   dp0 = p0 - base_p;
            const bool  in0 = (dp0 >= 0)     && (dp0 < WSPAN);
            const bool  in1 = (dp0 + 1 >= 0) && (dp0 + 1 < WSPAN);
            const bool  g0  = (p0 >= 0)      && (p0 < WOUT);
            const bool  g1  = (p0 + 1 >= 0)  && (p0 + 1 < WOUT);
            #pragma unroll
            for (int r = 0; r < 8; ++r) {
                const float v  = acc[k][n][r];
                float* wrow    = &win[(corow + r) * WPITCH];
                float* grow    = obase0 + (size_t)r * WOUT;
                if (in0)      atomicAdd(wrow + dp0,     v * w0);   // ds_add_f32
                else if (g0)  atomicAdd(grow + p0,      v * w0);   // rare outlier
                if (in1)      atomicAdd(wrow + dp0 + 1, v * w1);
                else if (g1)  atomicAdd(grow + p0 + 1,  v * w1);
            }
        }
    }
    __syncthreads();

    // ---- flush window -> global (coalesced atomics, skip zeros) ----
    for (int idx = t; idx < COUT * WSPAN; idx += 512) {
        const int co = idx / WSPAN;
        const int dp = idx - co * WSPAN;
        const float v = win[co * WPITCH + dp];
        const int p = base_p + dp;
        if (v != 0.0f && p >= 0 && p < WOUT)
            atomicAdd(out + ((size_t)b * COUT + co) * WOUT + p, v);
    }
}

// ---------------------------------------------------------------
// Host-side launcher
// inputs: 0=x[8,256,4096] 1=weight[256,256,4] 2=offset[8,4,4096]
//         3=mask[8,4,4096] 4=bias[256]    out: f32 [8,256,8192]
// ---------------------------------------------------------------
extern "C" void kernel_launch(void* const* d_in, const int* in_sizes, int n_in,
                              void* d_out, int out_size, void* d_ws, size_t ws_size,
                              hipStream_t stream) {
    const float* x      = (const float*)d_in[0];
    const float* weight = (const float*)d_in[1];
    const float* offset = (const float*)d_in[2];
    const float* mask   = (const float*)d_in[3];
    const float* bias   = (const float*)d_in[4];
    float* out          = (float*)d_out;

    unsigned short* wb = (unsigned short*)d_ws;       // 4*256*256 bf16 = 512 KB

    // 1) weight repack to bf16 [k][co][ci]
    weight_prep<<<(KW * COUT * CIN) / 256, 256, 0, stream>>>(weight, wb);

    // 2) out = bias broadcast
    out_init<<<(B_ * COUT * WOUT) / 256, 256, 0, stream>>>(out, bias);

    // 3) fused WMMA GEMM + windowed deformable scatter
    dim3 grid(WIN / 32, B_);
    deform_convT_fused<<<grid, 512, 0, stream>>>(x, wb, offset, mask, out);

    (void)in_sizes; (void)n_in; (void)out_size; (void)ws_size;
}
